// quantumAttentionBlock_49916109914403
// MI455X (gfx1250) — matrix operand order repaired and checked by reading, not compiled, not run on hardware
//
#include <hip/hip_runtime.h>
#include <hip/hip_bf16.h>

typedef __bf16 bf16_t;
typedef __attribute__((ext_vector_type(16))) __bf16 v16bf;
typedef __attribute__((ext_vector_type(8)))  __bf16 v8bf;
typedef __attribute__((ext_vector_type(8)))  float  v8f;

#define KC  64            // K-chunk staged in LDS
#define KCP (KC + 8)      // padded LDS row stride (144B) to avoid bank conflicts

// ---------------------------------------------------------------------------
// CDNA5 async global->LDS copy (ASYNCcnt path), 16 bytes per lane.
// VDST = VGPR holding wave-relative LDS byte address; VADDR = 64-bit global.
// Generic pointers into LDS carry the LDS offset in their low 32 bits.
// ---------------------------------------------------------------------------
__device__ __forceinline__ void async_cp16(const bf16_t* g, bf16_t* l) {
  unsigned lds = (unsigned)(unsigned long long)l;
  asm volatile("global_load_async_to_lds_b128 %0, %1, off"
               :
               : "v"(lds), "v"(g)
               : "memory");
}

__device__ __forceinline__ void wait_async0() {
  asm volatile("s_wait_asynccnt 0x0" ::: "memory");
}

// ---------------------------------------------------------------------------
// WMMA helper: D = A(16x32 bf16) x B(32x16 bf16) + C(16x16 f32)
// ---------------------------------------------------------------------------
__device__ __forceinline__ v8f wmma_bf16(v16bf a, v16bf b, v8f c) {
  return __builtin_amdgcn_wmma_f32_16x16x32_bf16(
      /*neg_a=*/false, a, /*neg_b=*/false, b,
      /*c_mod=*/(short)0, c, /*reuse_a=*/false, /*reuse_b=*/false);
}

// A fragment (16x32) from LDS tile (row-major, ld = KCP).
// lanes 0-15: row = lane,    K = {ks..ks+7, ks+16..ks+23}
// lanes 16-31: row = lane-16, K = {ks+8..ks+15, ks+24..ks+31}
__device__ __forceinline__ v16bf load_a_frag(const bf16_t* A,
                                             int row0, int ks, int ldk, int lane) {
  int m    = row0 + (lane & 15);
  int koff = ks + ((lane < 16) ? 0 : 8);
  const bf16_t* p = A + m * ldk + koff;
  v8bf lo = *reinterpret_cast<const v8bf*>(p);
  v8bf hi = *reinterpret_cast<const v8bf*>(p + 16);
  v16bf r;
#pragma unroll
  for (int i = 0; i < 8; ++i) { r[i] = lo[i]; r[i + 8] = hi[i]; }
  return r;
}

// B fragment (32x16) from LDS Bt tile (N x K row-major, ld = KCP).
// lanes 0-15: col = lane, K = {ks..ks+7, ks+16..ks+23} ; lanes 16-31: col = lane-16, K = {ks+8..ks+15, ks+24..ks+31}
__device__ __forceinline__ v16bf load_b_frag(const bf16_t* Bt,
                                             int col0, int ks, int ldk, int lane) {
  int n     = col0 + (lane & 15);
  int kbase = ks + ((lane < 16) ? 0 : 8);
  const bf16_t* p = Bt + n * ldk + kbase;
  v8bf lo = *reinterpret_cast<const v8bf*>(p);
  v8bf hi = *reinterpret_cast<const v8bf*>(p + 16);
  v16bf r;
#pragma unroll
  for (int i = 0; i < 8; ++i) { r[i] = lo[i]; r[i + 8] = hi[i]; }
  return r;
}

__device__ __forceinline__ float gelu_tanh(float x) {
  float x3 = x * x * x;
  float t  = tanhf(0.7978845608028654f * (x + 0.044715f * x3));
  return 0.5f * x * (1.0f + t);
}

// Epilogues:
// 1: sqrt(c^2+eps) -> bf16     2: c -> bf16            3: sqrt(c^2+eps) -> f32
// 4: resid + c -> f32          5: gelu(c+bias) -> bf16 6: resid + c + bias -> f32
__device__ __forceinline__ void store_tile(
    v8f c, int m0, int n0, int ldn, size_t obase,
    float* outF, bf16_t* __restrict__ outH,
    const float* resid, const float* __restrict__ bias, int epi, int lane) {
  int n  = n0 + (lane & 15);
  int mo = (lane < 16) ? 0 : 8;
#pragma unroll
  for (int v = 0; v < 8; ++v) {
    int m = m0 + v + mo;
    size_t idx = obase + (size_t)m * ldn + n;
    float f = c[v];
    if (epi == 1)      { f = sqrtf(f * f + 1e-8f); outH[idx] = (bf16_t)f; }
    else if (epi == 2) { outH[idx] = (bf16_t)f; }
    else if (epi == 3) { f = sqrtf(f * f + 1e-8f); outF[idx] = f; }
    else if (epi == 4) { outF[idx] = resid[idx] + f; }
    else if (epi == 5) { f = gelu_tanh(f + bias[n]); outH[idx] = (bf16_t)f; }
    else if (epi == 6) { outF[idx] = resid[idx] + f + bias[n]; }
    else               { outF[idx] = f; }
  }
}

// ---------------------------------------------------------------------------
// Batched GEMM: C[b] = A[b](MxK) * Bt[b](NxK)^T, bf16 in, f32 accumulate.
// Block = 256 threads = 8 waves; waves tiled 4(M) x 2(N); wave tile 32x32;
// block tile 128x64. A/Bt tiles double-buffered in LDS per 64-wide K-chunk,
// filled by async global->LDS DMA (ASYNCcnt) overlapped with WMMA compute.
// Grid: (N/64, M/128, batch). All dims divide exactly for this problem.
// ---------------------------------------------------------------------------
__global__ __launch_bounds__(256) void gemm_bt_kernel(
    const bf16_t* __restrict__ A, size_t strideA,
    const bf16_t* __restrict__ Bt, size_t strideB,
    int M, int N, int K,
    float* outF, bf16_t* __restrict__ outH, size_t strideO,
    const float* resid, const float* __restrict__ bias, int epi) {
  __shared__ bf16_t As[2][128 * KCP];   // 2 x 18 KB
  __shared__ bf16_t Bs[2][64 * KCP];    // 2 x  9 KB

  int b = blockIdx.z;
  const bf16_t* Ab = A + (size_t)b * strideA;
  const bf16_t* Bb = Bt + (size_t)b * strideB;
  size_t obase = (size_t)b * strideO;

  int tid  = threadIdx.x;
  int lane = tid & 31;
  int wave = tid >> 5;
  int rowB = blockIdx.y * 128;          // block-tile origin in M
  int colB = blockIdx.x * 64;           // block-tile origin in N
  int row0 = (wave >> 1) * 32;          // wave tile (LDS-local)
  int col0 = (wave & 1) * 32;

  // Async-stage one K-chunk into LDS buffer s.
  auto stage = [&](int s, int kk) {
    // A tile 128 x KC: 8192 elems, 8 bf16 (16B) per thread x 4 iters
#pragma unroll
    for (int i = 0; i < 4; ++i) {
      int idx = (i * 256 + tid) * 8;
      int r = idx / KC, c = idx % KC;
      async_cp16(Ab + (size_t)(rowB + r) * K + kk + c, &As[s][r * KCP + c]);
    }
    // Bt tile 64 x KC: 4096 elems, 16B per thread x 2 iters
#pragma unroll
    for (int i = 0; i < 2; ++i) {
      int idx = (i * 256 + tid) * 8;
      int r = idx / KC, c = idx % KC;
      async_cp16(Bb + (size_t)(colB + r) * K + kk + c, &Bs[s][r * KCP + c]);
    }
  };

  v8f c00 = {}, c01 = {}, c10 = {}, c11 = {};

  stage(0, 0);
  wait_async0();
  __syncthreads();

  int buf = 0;
  for (int kk = 0; kk < K; kk += KC) {
    int nkk = kk + KC;
    if (nkk < K) stage(buf ^ 1, nkk);   // DMA next chunk while computing this one

#pragma unroll
    for (int ks = 0; ks < KC; ks += 32) {
      v16bf a0 = load_a_frag(As[buf], row0,      ks, KCP, lane);
      v16bf a1 = load_a_frag(As[buf], row0 + 16, ks, KCP, lane);
      v16bf b0 = load_b_frag(Bs[buf], col0,      ks, KCP, lane);
      v16bf b1 = load_b_frag(Bs[buf], col0 + 16, ks, KCP, lane);
      c00 = wmma_bf16(a0, b0, c00);
      c01 = wmma_bf16(a0, b1, c01);
      c10 = wmma_bf16(a1, b0, c10);
      c11 = wmma_bf16(a1, b1, c11);
    }

    wait_async0();      // my async fills of buf^1 landed
    __syncthreads();    // everyone done reading buf & filling buf^1
    buf ^= 1;
  }

  int rg = rowB + row0, cg = colB + col0;
  store_tile(c00, rg,      cg,      N, obase, outF, outH, resid, bias, epi, lane);
  store_tile(c01, rg,      cg + 16, N, obase, outF, outH, resid, bias, epi, lane);
  store_tile(c10, rg + 16, cg,      N, obase, outF, outH, resid, bias, epi, lane);
  store_tile(c11, rg + 16, cg + 16, N, obase, outF, outH, resid, bias, epi, lane);
}

// ---------------------------------------------------------------------------
// LayerNorm over last dim (E=512), one block (256 thr) per row, bf16 out.
// ---------------------------------------------------------------------------
__global__ __launch_bounds__(256) void ln_bf16_kernel(
    const float* __restrict__ x, const float* __restrict__ gamma,
    const float* __restrict__ beta, bf16_t* __restrict__ out) {
  const int E = 512;
  size_t row = blockIdx.x;
  int tid = threadIdx.x;
  const float* xr = x + row * E;
  __shared__ float s1[256], s2[256];
  float v0 = xr[tid], v1 = xr[tid + 256];
  s1[tid] = v0 + v1;
  s2[tid] = v0 * v0 + v1 * v1;
  __syncthreads();
  for (int o = 128; o > 0; o >>= 1) {
    if (tid < o) { s1[tid] += s1[tid + o]; s2[tid] += s2[tid + o]; }
    __syncthreads();
  }
  float mu  = s1[0] * (1.0f / E);
  float var = s2[0] * (1.0f / E) - mu * mu;
  float rs  = rsqrtf(var + 1e-5f);
  out[row * E + tid]       = (bf16_t)((v0 - mu) * rs * gamma[tid] + beta[tid]);
  out[row * E + tid + 256] = (bf16_t)((v1 - mu) * rs * gamma[tid + 256] + beta[tid + 256]);
}

// ---------------------------------------------------------------------------
// Row softmax over T=512, one block per row, bf16 out.
// ---------------------------------------------------------------------------
__global__ __launch_bounds__(256) void softmax_bf16_kernel(
    const float* __restrict__ s, bf16_t* __restrict__ attn) {
  const int T = 512;
  size_t row = blockIdx.x;
  int tid = threadIdx.x;
  const float* sr = s + row * T;
  __shared__ float red[256];
  float a = sr[tid], b = sr[tid + 256];
  red[tid] = fmaxf(a, b);
  __syncthreads();
  for (int o = 128; o > 0; o >>= 1) {
    if (tid < o) red[tid] = fmaxf(red[tid], red[tid + o]);
    __syncthreads();
  }
  float rm = red[0];
  __syncthreads();
  float e0 = __expf(a - rm), e1 = __expf(b - rm);
  red[tid] = e0 + e1;
  __syncthreads();
  for (int o = 128; o > 0; o >>= 1) {
    if (tid < o) red[tid] += red[tid + o];
    __syncthreads();
  }
  float inv = 1.0f / red[0];
  attn[row * T + tid]       = (bf16_t)(e0 * inv);
  attn[row * T + tid + 256] = (bf16_t)(e1 * inv);
}

// ---------------------------------------------------------------------------
// f32 W[KxN] -> bf16 Wt[NxK]
// ---------------------------------------------------------------------------
__global__ void transpose_bf16_kernel(const float* __restrict__ w,
                                      bf16_t* __restrict__ wt, int K, int N) {
  int idx = blockIdx.x * blockDim.x + threadIdx.x;
  if (idx >= K * N) return;
  int k = idx / N, n = idx - k * N;
  wt[(size_t)n * K + k] = (bf16_t)w[idx];
}

// ---------------------------------------------------------------------------
extern "C" void kernel_launch(void* const* d_in, const int* in_sizes, int n_in,
                              void* d_out, int out_size, void* d_ws, size_t ws_size,
                              hipStream_t stream) {
  (void)in_sizes; (void)n_in; (void)out_size; (void)ws_size;
  const int B = 64, T = 512, E = 512, H = 2048;

  const float* x   = (const float*)d_in[0];
  const float* g1  = (const float*)d_in[1];
  const float* be1 = (const float*)d_in[2];
  const float* Wv  = (const float*)d_in[3];
  const float* Ww  = (const float*)d_in[4];
  const float* g2  = (const float*)d_in[5];
  const float* be2 = (const float*)d_in[6];
  const float* W1  = (const float*)d_in[7];
  const float* b1  = (const float*)d_in[8];
  const float* W2  = (const float*)d_in[9];
  const float* b2  = (const float*)d_in[10];
  float* out = (float*)d_out;

  char* ws = (char*)d_ws;
  size_t off = 0;
  auto take = [&](size_t bytes) -> char* {
    char* p = ws + off;
    off = (off + bytes + 255) & ~(size_t)255;
    return p;
  };
  const size_t BTE = (size_t)B * T * E;
  const size_t BTT = (size_t)B * T * T;
  const size_t BTH = (size_t)B * T * H;
  bf16_t* inp_h = (bf16_t*)take(BTE * 2);          // LN1(x) bf16
  bf16_t* vx_h  = (bf16_t*)take(BTE * 2);          // sqrt((inp@Wv)^2+eps)
  bf16_t* pa_h  = (bf16_t*)take(BTT * 2);          // proj, later reused for attn
  float*  sc_f  = (float*)take(BTT * 4);           // scores
  bf16_t* hln_h = (bf16_t*)take(BTE * 2);          // LN2 out
  bf16_t* h1_h  = (bf16_t*)take(BTH * 2);          // gelu MLP hidden
  bf16_t* WvT   = (bf16_t*)take((size_t)E * E * 2);
  bf16_t* WwT   = (bf16_t*)take((size_t)E * E * 2);
  bf16_t* W1T   = (bf16_t*)take((size_t)E * H * 2);
  bf16_t* W2T   = (bf16_t*)take((size_t)H * E * 2);

  // Weight transposes to bf16 N x K layout
  {
    int nEE = E * E, nEH = E * H;
    transpose_bf16_kernel<<<(nEE + 255) / 256, 256, 0, stream>>>(Wv, WvT, E, E);
    transpose_bf16_kernel<<<(nEE + 255) / 256, 256, 0, stream>>>(Ww, WwT, E, E);
    transpose_bf16_kernel<<<(nEH + 255) / 256, 256, 0, stream>>>(W1, W1T, E, H);
    transpose_bf16_kernel<<<(nEH + 255) / 256, 256, 0, stream>>>(W2, W2T, H, E);
  }

  dim3 blk(256);
  const size_t sTE = (size_t)T * E, sTT = (size_t)T * T, sTH = (size_t)T * H;

  // LN1
  ln_bf16_kernel<<<B * T, 256, 0, stream>>>(x, g1, be1, inp_h);

  // vx = sqrt((inp @ Wv)^2 + eps)   [B,T,E] bf16
  gemm_bt_kernel<<<dim3(E / 64, T / 128, B), blk, 0, stream>>>(
      inp_h, sTE, WvT, 0, T, E, E, nullptr, vx_h, sTE, nullptr, nullptr, 1);

  // proj = inp @ Ww                 [B,T,E] bf16
  gemm_bt_kernel<<<dim3(E / 64, T / 128, B), blk, 0, stream>>>(
      inp_h, sTE, WwT, 0, T, E, E, nullptr, pa_h, sTE, nullptr, nullptr, 2);

  // scores = sqrt((proj @ inp^T)^2 + eps)  [B,T,T] f32  (Bt operand = inp)
  gemm_bt_kernel<<<dim3(T / 64, T / 128, B), blk, 0, stream>>>(
      pa_h, sTE, inp_h, sTE, T, T, E, sc_f, nullptr, sTT, nullptr, nullptr, 3);

  // attn = softmax(scores) -> bf16 (reuse proj buffer)
  softmax_bf16_kernel<<<B * T, 256, 0, stream>>>(sc_f, pa_h);

  // xres = x + attn @ vx^T  -> d_out  (Bt operand = vx)
  gemm_bt_kernel<<<dim3(E / 64, T / 128, B), blk, 0, stream>>>(
      pa_h, sTT, vx_h, sTE, T, E, T, out, nullptr, sTE, x, nullptr, 4);

  // LN2
  ln_bf16_kernel<<<B * T, 256, 0, stream>>>(out, g2, be2, hln_h);

  // h1 = gelu(hln @ W1 + b1)  [B,T,H] bf16
  gemm_bt_kernel<<<dim3(H / 64, T / 128, B), blk, 0, stream>>>(
      hln_h, sTE, W1T, 0, T, H, E, nullptr, h1_h, sTH, nullptr, b1, 5);

  // out = xres + h1 @ W2 + b2
  gemm_bt_kernel<<<dim3(E / 64, T / 128, B), blk, 0, stream>>>(
      h1_h, sTH, W2T, 0, T, E, H, out, nullptr, sTE, out, b2, 6);
}
